// TransforMatcher_73048803770572
// MI455X (gfx1250) — matrix operand order repaired and checked by reading, not compiled, not run on hardware
//
#include <hip/hip_runtime.h>
#include <math.h>

#define N_TOK 810000
#define PI_F  3.14159265358979f
#define PI5_F 15.7079632679490f

typedef __attribute__((ext_vector_type(16))) _Float16 v16h;
typedef __attribute__((ext_vector_type(8)))  _Float16 v8h;
typedef __attribute__((ext_vector_type(8)))  float    v8f;

// ---------------- small device helpers ----------------

__device__ __forceinline__ void ln6(const float* xr, const float* g, const float* b, float* o) {
  float m = 0.f;
  for (int d = 0; d < 6; ++d) m += xr[d];
  m *= (1.f / 6.f);
  float v = 0.f;
  for (int d = 0; d < 6; ++d) { float t = xr[d] - m; v += t * t; }
  v *= (1.f / 6.f);
  float r = rsqrtf(v + 1e-5f);
  for (int d = 0; d < 6; ++d) o[d] = (xr[d] - m) * r * g[d] + b[d];
}

// xn[6] @ W[6][96], columns [c0, c0+32)
__device__ __forceinline__ void matvec6(const float* xn, const float* W, int c0, float* out) {
  for (int j = 0; j < 32; ++j) {
    float a = 0.f;
    for (int d = 0; d < 6; ++d) a += xn[d] * W[d * 96 + c0 + j];
    out[j] = a;
  }
}

__device__ __forceinline__ void atomicMaxF(float* addr, float val) {
  unsigned int* ai = (unsigned int*)addr;
  unsigned int old = *ai;
  while (__uint_as_float(old) < val) {
    unsigned int assumed = old;
    old = atomicCAS(ai, assumed, __float_as_uint(val));
    if (old == assumed) break;
  }
}

__device__ __forceinline__ float waveReduceAdd(float v) {
  for (int o = 16; o; o >>= 1) v += __shfl_down(v, o, 32);
  return v;
}
__device__ __forceinline__ float waveReduceMax(float v) {
  for (int o = 16; o; o >>= 1) v = fmaxf(v, __shfl_down(v, o, 32));
  return v;
}

__device__ __forceinline__ v16h makeA(const _Float16* arow, int kk) {
  v16h av = *(const v16h*)(arow + kk);
  return av;
}

// ---------------- phase 1: maxpool2 + channel L2 norm -> f16 features [4][960][256] (pos-major)

__global__ void pool_norm_kernel(const float* __restrict__ src0, const float* __restrict__ src1,
                                 const float* __restrict__ trg0, const float* __restrict__ trg1,
                                 _Float16* __restrict__ feat) {
  int p = blockIdx.x;       // 0..959 (900 real, 60 zero pad rows)
  int which = blockIdx.y;   // 0:sn0 1:tn0 2:sn1 3:tn1
  const float* s = (which == 0) ? src0 : (which == 1) ? trg0 : (which == 2) ? src1 : trg1;
  int c = threadIdx.x;      // 256 channels
  float m = 0.f;
  if (p < 900) {
    int y = p / 30, xx = p % 30;
    const float* base = s + (size_t)c * 3600 + (2 * y) * 60 + 2 * xx;
    m = fmaxf(fmaxf(base[0], base[1]), fmaxf(base[60], base[61]));
  }
  __shared__ float red[256];
  red[c] = m * m;
  __syncthreads();
  for (int o = 128; o; o >>= 1) { if (c < o) red[c] += red[c + o]; __syncthreads(); }
  float rn = rsqrtf(red[0] + 1e-6f);
  feat[((size_t)which * 960 + p) * 256 + c] = (_Float16)(m * rn);
}

// ---------------- phase 2: correlation GEMM 900x900x256 per pair, ReLU, via WMMA f16
// each wave: 16(p) x 64(q) output tile, A fragment reused across 4 WMMAs per k-step

__global__ void corr_wmma_kernel(const _Float16* __restrict__ feat, float* __restrict__ corrbuf) {
  int pair = blockIdx.z;
  const _Float16* A  = feat + (size_t)(pair * 2 + 0) * 960 * 256;  // sn^T [p][c]
  const _Float16* Bm = feat + (size_t)(pair * 2 + 1) * 960 * 256;  // tn^T [q][c]
  int wave = threadIdx.x >> 5, lane = threadIdx.x & 31;
  int qg = blockIdx.x * 8 + wave;       // q-group of 64
  if (qg >= 15) return;
  int ptile = blockIdx.y * 16, qbase = qg * 64;
  int half = lane >> 4, l15 = lane & 15;
  const _Float16* arow = A  + (size_t)(ptile + l15) * 256 + half * 16;
  const _Float16* bcol = Bm + (size_t)(qbase + l15) * 256 + half * 16;
  const size_t bs = (size_t)16 * 256;
  v8f acc0 = {}, acc1 = {}, acc2 = {}, acc3 = {};
  for (int kk = 0; kk < 256; kk += 32) {
    v16h av = makeA(arow, kk);
    v16h b0 = *(const v16h*)(bcol + kk);
    v16h b1 = *(const v16h*)(bcol + bs + kk);
    v16h b2 = *(const v16h*)(bcol + 2 * bs + kk);
    v16h b3 = *(const v16h*)(bcol + 3 * bs + kk);
    acc0 = __builtin_amdgcn_wmma_f32_16x16x32_f16(false, av, false, b0, (short)0, acc0, false, false);
    acc1 = __builtin_amdgcn_wmma_f32_16x16x32_f16(false, av, false, b1, (short)0, acc1, false, false);
    acc2 = __builtin_amdgcn_wmma_f32_16x16x32_f16(false, av, false, b2, (short)0, acc2, false, false);
    acc3 = __builtin_amdgcn_wmma_f32_16x16x32_f16(false, av, false, b3, (short)0, acc3, false, false);
  }
  float* out = corrbuf + (size_t)pair * N_TOK;
  int prow = ptile + half * 8;
  for (int t = 0; t < 4; ++t) {
    const v8f& a = (t == 0) ? acc0 : (t == 1) ? acc1 : (t == 2) ? acc2 : acc3;
    int q = qbase + t * 16 + l15;
    if (q < 900) {
      for (int r = 0; r < 8; ++r) {
        int p = prow + r;
        if (p < 900) out[(size_t)p * 900 + q] = fmaxf(a[r], 0.f);
      }
    }
  }
}

// ---------------- phase 3: projection to x [N][6]

__global__ void proj_kernel(const float* __restrict__ corrbuf, const float* __restrict__ proj_w,
                            const float* __restrict__ proj_b, float* __restrict__ x) {
  size_t n = (size_t)blockIdx.x * blockDim.x + threadIdx.x;
  if (n >= N_TOK) return;
  float c0 = corrbuf[n], c1 = corrbuf[N_TOK + n];
  for (int d = 0; d < 6; ++d)
    x[n * 6 + d] = c0 * proj_w[d] + c1 * proj_w[6 + d] + proj_b[d];
}

// ---------------- transformer layer kernels
// acc layout (floats): [0..7] qmax  [8..15] qden  [16..47] gqn  [48..79] gq
//                      [80..87] kmax [88..95] kden [96..127] gkn [128..159] gk

__global__ void init_acc_kernel(float* __restrict__ acc) {
  int t = threadIdx.x;
  if (t < 160) acc[t] = ((t < 8) || (t >= 80 && t < 88)) ? -INFINITY : 0.f;
}

__global__ void qmax_kernel(const float* __restrict__ x, const float* __restrict__ qkv_w,
                            const float* __restrict__ ln1_g, const float* __restrict__ ln1_b,
                            const float* __restrict__ qattn_w, int layer, float* __restrict__ acc) {
  __shared__ float W[576], g[6], b[6], qw[4];
  const float* Wl = qkv_w + (size_t)layer * 576;
  for (int t = threadIdx.x; t < 576; t += blockDim.x) W[t] = Wl[t];
  if (threadIdx.x < 6) { g[threadIdx.x] = ln1_g[layer * 6 + threadIdx.x]; b[threadIdx.x] = ln1_b[layer * 6 + threadIdx.x]; }
  if (threadIdx.x < 4) qw[threadIdx.x] = qattn_w[layer * 4 + threadIdx.x];
  __syncthreads();
  float lmax[8];
  for (int h = 0; h < 8; ++h) lmax[h] = -INFINITY;
  for (size_t n = (size_t)blockIdx.x * blockDim.x + threadIdx.x; n < N_TOK;
       n += (size_t)gridDim.x * blockDim.x) {
    float xr[6], xn[6], q[32];
    for (int d = 0; d < 6; ++d) xr[d] = x[n * 6 + d];
    ln6(xr, g, b, xn);
    matvec6(xn, W, 0, q);
    for (int h = 0; h < 8; ++h) {
      const float* qh = q + 4 * h;
      float l = (qh[0] * qw[0] + qh[1] * qw[1] + qh[2] * qw[2] + qh[3] * qw[3]) * 0.5f;
      lmax[h] = fmaxf(lmax[h], l);
    }
  }
  int lane = threadIdx.x & 31;
  for (int h = 0; h < 8; ++h) {
    float v = waveReduceMax(lmax[h]);
    if (lane == 0) atomicMaxF(&acc[h], v);
  }
}

__global__ void qsum_kernel(const float* __restrict__ x, const float* __restrict__ qkv_w,
                            const float* __restrict__ ln1_g, const float* __restrict__ ln1_b,
                            const float* __restrict__ qattn_w, int layer, float* __restrict__ acc) {
  __shared__ float W[576], g[6], b[6], qw[4], qmaxs[8];
  const float* Wl = qkv_w + (size_t)layer * 576;
  for (int t = threadIdx.x; t < 576; t += blockDim.x) W[t] = Wl[t];
  if (threadIdx.x < 6) { g[threadIdx.x] = ln1_g[layer * 6 + threadIdx.x]; b[threadIdx.x] = ln1_b[layer * 6 + threadIdx.x]; }
  if (threadIdx.x < 4) qw[threadIdx.x] = qattn_w[layer * 4 + threadIdx.x];
  if (threadIdx.x < 8) qmaxs[threadIdx.x] = acc[threadIdx.x];
  __syncthreads();
  float den[8], num[32];
  for (int h = 0; h < 8; ++h) den[h] = 0.f;
  for (int j = 0; j < 32; ++j) num[j] = 0.f;
  for (size_t n = (size_t)blockIdx.x * blockDim.x + threadIdx.x; n < N_TOK;
       n += (size_t)gridDim.x * blockDim.x) {
    float xr[6], xn[6], q[32];
    for (int d = 0; d < 6; ++d) xr[d] = x[n * 6 + d];
    ln6(xr, g, b, xn);
    matvec6(xn, W, 0, q);
    float fn = (float)n, sA, cA, sB, cB;
    sincosf(fn * PI_F, &sA, &cA);
    sincosf(fn * PI5_F, &sB, &cB);
    for (int h = 0; h < 8; ++h) {
      const float* qh = q + 4 * h;
      float l = (qh[0] * qw[0] + qh[1] * qw[1] + qh[2] * qw[2] + qh[3] * qw[3]) * 0.5f;
      float e = __expf(l - qmaxs[h]);
      den[h] += e;
      num[4 * h + 0] += e * (qh[0] * cA - qh[1] * sA);
      num[4 * h + 1] += e * (qh[1] * cA + qh[0] * sA);
      num[4 * h + 2] += e * (qh[2] * cB - qh[3] * sB);
      num[4 * h + 3] += e * (qh[3] * cB + qh[2] * sB);
    }
  }
  int lane = threadIdx.x & 31;
  for (int h = 0; h < 8; ++h) {
    float v = waveReduceAdd(den[h]);
    if (lane == 0) atomicAdd(&acc[8 + h], v);
  }
  for (int j = 0; j < 32; ++j) {
    float v = waveReduceAdd(num[j]);
    if (lane == 0) atomicAdd(&acc[16 + j], v);
  }
}

__global__ void finalize_gq_kernel(float* __restrict__ acc) {
  int t = threadIdx.x;
  if (t < 32) acc[48 + t] = acc[16 + t] / acc[8 + (t >> 2)];
}

__global__ void kmax_kernel(const float* __restrict__ x, const float* __restrict__ qkv_w,
                            const float* __restrict__ ln1_g, const float* __restrict__ ln1_b,
                            const float* __restrict__ kattn_w, int layer, float* __restrict__ acc) {
  __shared__ float W[576], g[6], b[6], kw[2], gqs[32];
  const float* Wl = qkv_w + (size_t)layer * 576;
  for (int t = threadIdx.x; t < 576; t += blockDim.x) W[t] = Wl[t];
  if (threadIdx.x < 6) { g[threadIdx.x] = ln1_g[layer * 6 + threadIdx.x]; b[threadIdx.x] = ln1_b[layer * 6 + threadIdx.x]; }
  if (threadIdx.x < 2) kw[threadIdx.x] = kattn_w[layer * 2 + threadIdx.x];
  if (threadIdx.x < 32) gqs[threadIdx.x] = acc[48 + threadIdx.x];
  __syncthreads();
  float lmax[8];
  for (int h = 0; h < 8; ++h) lmax[h] = -INFINITY;
  for (size_t n = (size_t)blockIdx.x * blockDim.x + threadIdx.x; n < N_TOK;
       n += (size_t)gridDim.x * blockDim.x) {
    float xr[6], xn[6], k[32];
    for (int d = 0; d < 6; ++d) xr[d] = x[n * 6 + d];
    ln6(xr, g, b, xn);
    matvec6(xn, W, 32, k);
    for (int h = 0; h < 8; ++h) {
      const float* kh = k + 4 * h;
      float k20 = kh[0] * gqs[4 * h + 0] + kh[1] * gqs[4 * h + 1];
      float k21 = kh[2] * gqs[4 * h + 2] + kh[3] * gqs[4 * h + 3];
      float l = (k20 * kw[0] + k21 * kw[1]) * 0.5f;
      lmax[h] = fmaxf(lmax[h], l);
    }
  }
  int lane = threadIdx.x & 31;
  for (int h = 0; h < 8; ++h) {
    float v = waveReduceMax(lmax[h]);
    if (lane == 0) atomicMaxF(&acc[80 + h], v);
  }
}

__global__ void ksum_kernel(const float* __restrict__ x, const float* __restrict__ qkv_w,
                            const float* __restrict__ ln1_g, const float* __restrict__ ln1_b,
                            const float* __restrict__ kattn_w, int layer, float* __restrict__ acc) {
  __shared__ float W[576], g[6], b[6], kw[2], gqs[32], kmaxs[8];
  const float* Wl = qkv_w + (size_t)layer * 576;
  for (int t = threadIdx.x; t < 576; t += blockDim.x) W[t] = Wl[t];
  if (threadIdx.x < 6) { g[threadIdx.x] = ln1_g[layer * 6 + threadIdx.x]; b[threadIdx.x] = ln1_b[layer * 6 + threadIdx.x]; }
  if (threadIdx.x < 2) kw[threadIdx.x] = kattn_w[layer * 2 + threadIdx.x];
  if (threadIdx.x < 32) gqs[threadIdx.x] = acc[48 + threadIdx.x];
  if (threadIdx.x < 8)  kmaxs[threadIdx.x] = acc[80 + threadIdx.x];
  __syncthreads();
  float den[8], num[32];
  for (int h = 0; h < 8; ++h) den[h] = 0.f;
  for (int j = 0; j < 32; ++j) num[j] = 0.f;
  for (size_t n = (size_t)blockIdx.x * blockDim.x + threadIdx.x; n < N_TOK;
       n += (size_t)gridDim.x * blockDim.x) {
    float xr[6], xn[6], k[32];
    for (int d = 0; d < 6; ++d) xr[d] = x[n * 6 + d];
    ln6(xr, g, b, xn);
    matvec6(xn, W, 32, k);
    float fn = (float)n, sA, cA, sB, cB;
    sincosf(fn * PI_F, &sA, &cA);
    sincosf(fn * PI5_F, &sB, &cB);
    for (int h = 0; h < 8; ++h) {
      const float* kh = k + 4 * h;
      float k20 = kh[0] * gqs[4 * h + 0] + kh[1] * gqs[4 * h + 1];
      float k21 = kh[2] * gqs[4 * h + 2] + kh[3] * gqs[4 * h + 3];
      float l = (k20 * kw[0] + k21 * kw[1]) * 0.5f;
      float e = __expf(l - kmaxs[h]);
      den[h] += e;
      num[4 * h + 0] += e * (kh[0] * cA - kh[1] * sA);
      num[4 * h + 1] += e * (kh[1] * cA + kh[0] * sA);
      num[4 * h + 2] += e * (kh[2] * cB - kh[3] * sB);
      num[4 * h + 3] += e * (kh[3] * cB + kh[2] * sB);
    }
  }
  int lane = threadIdx.x & 31;
  for (int h = 0; h < 8; ++h) {
    float v = waveReduceAdd(den[h]);
    if (lane == 0) atomicAdd(&acc[88 + h], v);
  }
  for (int j = 0; j < 32; ++j) {
    float v = waveReduceAdd(num[j]);
    if (lane == 0) atomicAdd(&acc[96 + j], v);
  }
}

__global__ void finalize_gk_kernel(float* __restrict__ acc) {
  int t = threadIdx.x;
  if (t < 32) acc[128 + t] = acc[96 + t] / acc[88 + (t >> 2)];
}

__global__ void update_kernel(float* __restrict__ x, const float* __restrict__ qkv_w,
                              const float* __restrict__ ln1_g, const float* __restrict__ ln1_b,
                              const float* __restrict__ tor_w, const float* __restrict__ tor_b,
                              const float* __restrict__ out_w, const float* __restrict__ out_b,
                              const float* __restrict__ ln2_g, const float* __restrict__ ln2_b,
                              const float* __restrict__ ff1_w, const float* __restrict__ ff1_b,
                              const float* __restrict__ ff2_w, const float* __restrict__ ff2_b,
                              int layer, const float* __restrict__ acc) {
  __shared__ float W[576], g[6], b[6], torw[8], torb[4], outw[192], outb[6];
  __shared__ float g2[6], b2[6], f1w[144], f1b[24], f2w[144], f2b[6], gks[32];
  int tid = threadIdx.x;
  const float* Wl = qkv_w + (size_t)layer * 576;
  for (int t = tid; t < 576; t += blockDim.x) W[t] = Wl[t];
  for (int t = tid; t < 192; t += blockDim.x) outw[t] = out_w[layer * 192 + t];
  for (int t = tid; t < 144; t += blockDim.x) { f1w[t] = ff1_w[layer * 144 + t]; f2w[t] = ff2_w[layer * 144 + t]; }
  if (tid < 6) {
    g[tid] = ln1_g[layer * 6 + tid];  b[tid] = ln1_b[layer * 6 + tid];
    g2[tid] = ln2_g[layer * 6 + tid]; b2[tid] = ln2_b[layer * 6 + tid];
    outb[tid] = out_b[layer * 6 + tid]; f2b[tid] = ff2_b[layer * 6 + tid];
  }
  if (tid < 8)  torw[tid] = tor_w[layer * 8 + tid];
  if (tid < 4)  torb[tid] = tor_b[layer * 4 + tid];
  if (tid < 24) f1b[tid] = ff1_b[layer * 24 + tid];
  if (tid < 32) gks[tid] = acc[128 + tid];
  __syncthreads();
  size_t n = (size_t)blockIdx.x * blockDim.x + tid;
  if (n >= N_TOK) return;
  float xr[6], xn[6], q[32], vv[32];
  for (int d = 0; d < 6; ++d) xr[d] = x[n * 6 + d];
  ln6(xr, g, b, xn);
  matvec6(xn, W, 0, q);
  matvec6(xn, W, 64, vv);
  float y[6];
  for (int d = 0; d < 6; ++d) y[d] = xr[d] + outb[d];
  for (int h = 0; h < 8; ++h) {
    float u0 = vv[4 * h + 0] * gks[4 * h + 0] + vv[4 * h + 1] * gks[4 * h + 1];
    float u1 = vv[4 * h + 2] * gks[4 * h + 2] + vv[4 * h + 3] * gks[4 * h + 3];
    for (int d = 0; d < 4; ++d) {
      float r = u0 * torw[d] + u1 * torw[4 + d] + torb[d] + q[4 * h + d];
      for (int dd = 0; dd < 6; ++dd) y[dd] += r * outw[(h * 4 + d) * 6 + dd];
    }
  }
  float xn2[6];
  ln6(y, g2, b2, xn2);
  float hdn[24];
  for (int j = 0; j < 24; ++j) {
    float a = f1b[j];
    for (int d = 0; d < 6; ++d) a += xn2[d] * f1w[d * 24 + j];
    hdn[j] = 0.5f * a * (1.f + erff(a * 0.70710678f));
  }
  for (int d = 0; d < 6; ++d) {
    float a = f2b[d];
    for (int j = 0; j < 24; ++j) a += hdn[j] * f2w[j * 6 + d];
    x[n * 6 + d] = y[d] + a;
  }
}

// ---------------- phase 4: rc = x . corr_w + corr_b

__global__ void rc_kernel(const float* __restrict__ x, const float* __restrict__ corr_w,
                          const float* __restrict__ corr_b, float* __restrict__ rc30) {
  size_t n = (size_t)blockIdx.x * blockDim.x + threadIdx.x;
  if (n >= N_TOK) return;
  float a = corr_b[0];
  for (int d = 0; d < 6; ++d) a += x[n * 6 + d] * corr_w[d];
  rc30[n] = a;
}

// ---------------- phase 5: fused 4D interpolation + row softmax -> attn f16 [3600][3616]

__global__ void attn_softmax_kernel(const float* __restrict__ rc30, _Float16* __restrict__ attn) {
  int p = blockIdx.x, tid = threadIdx.x;
  int a = p / 60, bq = p % 60;
  float ca = a  * (29.f / 59.f); int ia0 = (int)floorf(ca); float wa = ca - ia0; int ia1 = min(ia0 + 1, 29);
  float cb = bq * (29.f / 59.f); int ib0 = (int)floorf(cb); float wb = cb - ib0; int ib1 = min(ib0 + 1, 29);
  float w00 = (1.f - wa) * (1.f - wb), w01 = (1.f - wa) * wb, w10 = wa * (1.f - wb), w11 = wa * wb;
  __shared__ float P[900];
  __shared__ float red[256];
  for (int t = tid; t < 900; t += blockDim.x)
    P[t] = w00 * rc30[(ia0 * 30 + ib0) * 900 + t] + w01 * rc30[(ia0 * 30 + ib1) * 900 + t] +
           w10 * rc30[(ia1 * 30 + ib0) * 900 + t] + w11 * rc30[(ia1 * 30 + ib1) * 900 + t];
  __syncthreads();
  float vals[15];
  float lmax = -INFINITY;
  for (int j = 0; j < 15; ++j) {
    int q = tid + j * 256;
    if (q < 3600) {
      int cq = q / 60, dq = q % 60;
      float cc = cq * (29.f / 59.f); int ic0 = (int)floorf(cc); float wc = cc - ic0; int ic1 = min(ic0 + 1, 29);
      float cd = dq * (29.f / 59.f); int id0 = (int)floorf(cd); float wd = cd - id0; int id1 = min(id0 + 1, 29);
      float v = (1.f - wc) * ((1.f - wd) * P[ic0 * 30 + id0] + wd * P[ic0 * 30 + id1]) +
                       wc  * ((1.f - wd) * P[ic1 * 30 + id0] + wd * P[ic1 * 30 + id1]);
      v *= 20.0f;  // TEMP
      vals[j] = v;
      lmax = fmaxf(lmax, v);
    }
  }
  red[tid] = lmax; __syncthreads();
  for (int o = 128; o; o >>= 1) { if (tid < o) red[tid] = fmaxf(red[tid], red[tid + o]); __syncthreads(); }
  float rowmax = red[0];
  __syncthreads();
  float lsum = 0.f;
  for (int j = 0; j < 15; ++j) {
    int q = tid + j * 256;
    if (q < 3600) { vals[j] = __expf(vals[j] - rowmax); lsum += vals[j]; }
  }
  red[tid] = lsum; __syncthreads();
  for (int o = 128; o; o >>= 1) { if (tid < o) red[tid] += red[tid + o]; __syncthreads(); }
  float inv = 1.f / red[0];
  for (int j = 0; j < 15; ++j) {
    int q = tid + j * 256;
    if (q < 3600) attn[(size_t)p * 3616 + q] = (_Float16)(vals[j] * inv);
  }
  if (tid < 16) attn[(size_t)p * 3616 + 3600 + tid] = (_Float16)0.f;
}

// ---------------- phase 6: v -> f16 K-padded [256][3616]

__global__ void vh_kernel(const float* __restrict__ v, _Float16* __restrict__ vh) {
  int idx = blockIdx.x * 256 + threadIdx.x;
  if (idx >= 256 * 3616) return;
  int c = idx / 3616, k = idx % 3616;
  vh[idx] = (_Float16)(k < 3600 ? v[(size_t)c * 3600 + k] : 0.f);
}

// ---------------- phase 7: wv = attn @ v^T via WMMA f16 (M=3600, N=256, K=3616)
// each wave: 16(p) x 64(c) output tile, A fragment reused across 4 WMMAs per k-step

__global__ void wv_wmma_kernel(const _Float16* __restrict__ attn, const _Float16* __restrict__ vh,
                               float* __restrict__ wv) {
  int wave = threadIdx.x >> 5, lane = threadIdx.x & 31;
  int tile = blockIdx.x * 8 + wave;   // 113 blocks * 8 waves >= 900 tiles
  if (tile >= 900) return;
  int mt = tile >> 2, ng = tile & 3;  // mt: 0..224, ng: group of 4 c-tiles
  int ptile = mt * 16, cbase = ng * 64;
  int half = lane >> 4, l15 = lane & 15;
  const _Float16* arow = attn + (size_t)(ptile + l15) * 3616 + half * 16;
  const _Float16* bcol = vh   + (size_t)(cbase + l15) * 3616 + half * 16;
  const size_t bs = (size_t)16 * 3616;
  v8f acc0 = {}, acc1 = {}, acc2 = {}, acc3 = {};
  for (int kk = 0; kk < 3616; kk += 32) {
    __builtin_prefetch(arow + kk + 128, 0, 0);
    v16h av = makeA(arow, kk);
    v16h b0 = *(const v16h*)(bcol + kk);
    v16h b1 = *(const v16h*)(bcol + bs + kk);
    v16h b2 = *(const v16h*)(bcol + 2 * bs + kk);
    v16h b3 = *(const v16h*)(bcol + 3 * bs + kk);
    acc0 = __builtin_amdgcn_wmma_f32_16x16x32_f16(false, av, false, b0, (short)0, acc0, false, false);
    acc1 = __builtin_amdgcn_wmma_f32_16x16x32_f16(false, av, false, b1, (short)0, acc1, false, false);
    acc2 = __builtin_amdgcn_wmma_f32_16x16x32_f16(false, av, false, b2, (short)0, acc2, false, false);
    acc3 = __builtin_amdgcn_wmma_f32_16x16x32_f16(false, av, false, b3, (short)0, acc3, false, false);
  }
  for (int t = 0; t < 4; ++t) {
    const v8f& a = (t == 0) ? acc0 : (t == 1) ? acc1 : (t == 2) ? acc2 : acc3;
    int c = cbase + t * 16 + l15;
    for (int r = 0; r < 8; ++r) {
      int p = ptile + r + half * 8;
      wv[(size_t)c * 3600 + p] = a[r];
    }
  }
}

// ---------------- phase 8: fq = normalize(f_q) + normalize(wv)*0.5

__global__ void final_kernel(const float* __restrict__ f_q, const float* __restrict__ wv,
                             float* __restrict__ fq_out) {
  int p = blockIdx.x, c = threadIdx.x;
  float a = f_q[(size_t)c * 3600 + p];
  float w = wv[(size_t)c * 3600 + p];
  __shared__ float r1[256], r2[256];
  r1[c] = a * a; r2[c] = w * w;
  __syncthreads();
  for (int o = 128; o; o >>= 1) { if (c < o) { r1[c] += r1[c + o]; r2[c] += r2[c + o]; } __syncthreads(); }
  float na = fmaxf(sqrtf(r1[0]), 1e-12f);
  float nw = fmaxf(sqrtf(r2[0]), 1e-12f);
  fq_out[(size_t)c * 3600 + p] = a / na + (w / nw) * 0.5f;
}

// ---------------- host ----------------

static inline size_t alignUp(size_t v) { return (v + 255) & ~(size_t)255; }

extern "C" void kernel_launch(void* const* d_in, const int* in_sizes, int n_in,
                              void* d_out, int out_size, void* d_ws, size_t ws_size,
                              hipStream_t stream) {
  const float* src0   = (const float*)d_in[0];
  const float* src1   = (const float*)d_in[1];
  const float* trg0   = (const float*)d_in[2];
  const float* trg1   = (const float*)d_in[3];
  const float* f_q    = (const float*)d_in[4];
  const float* v      = (const float*)d_in[5];
  const float* proj_w = (const float*)d_in[6];
  const float* proj_b = (const float*)d_in[7];
  const float* ln1_g  = (const float*)d_in[8];
  const float* ln1_b  = (const float*)d_in[9];
  const float* qkv_w  = (const float*)d_in[10];
  const float* qattn_w= (const float*)d_in[11];
  const float* kattn_w= (const float*)d_in[12];
  const float* tor_w  = (const float*)d_in[13];
  const float* tor_b  = (const float*)d_in[14];
  const float* out_w  = (const float*)d_in[15];
  const float* out_b  = (const float*)d_in[16];
  const float* ln2_g  = (const float*)d_in[17];
  const float* ln2_b  = (const float*)d_in[18];
  const float* ff1_w  = (const float*)d_in[19];
  const float* ff1_b  = (const float*)d_in[20];
  const float* ff2_w  = (const float*)d_in[21];
  const float* ff2_b  = (const float*)d_in[22];
  const float* corr_w = (const float*)d_in[23];
  const float* corr_b = (const float*)d_in[24];

  char* wsb = (char*)d_ws;
  size_t off = 0;
  _Float16* feat = (_Float16*)(wsb + off); off = alignUp(off + (size_t)4 * 960 * 256 * 2);
  float* corrbuf = (float*)(wsb + off);    off = alignUp(off + (size_t)2 * N_TOK * 4);
  float* x       = (float*)(wsb + off);    off = alignUp(off + (size_t)N_TOK * 6 * 4);
  float* acc     = (float*)(wsb + off);    off = alignUp(off + 1024);
  float* rc30    = (float*)(wsb + off);    off = alignUp(off + (size_t)N_TOK * 4);
  _Float16* attn = (_Float16*)(wsb + off); off = alignUp(off + (size_t)3600 * 3616 * 2);
  _Float16* vh   = (_Float16*)(wsb + off); off = alignUp(off + (size_t)256 * 3616 * 2);

  float* fq_out = (float*)d_out;
  float* wv_out = (float*)d_out + 921600;

  const int tokBlocks = (N_TOK + 255) / 256;

  pool_norm_kernel<<<dim3(960, 4), 256, 0, stream>>>(src0, src1, trg0, trg1, feat);
  corr_wmma_kernel<<<dim3(2, 57, 2), 256, 0, stream>>>(feat, corrbuf);
  proj_kernel<<<tokBlocks, 256, 0, stream>>>(corrbuf, proj_w, proj_b, x);

  for (int l = 0; l < 6; ++l) {
    init_acc_kernel<<<1, 256, 0, stream>>>(acc);
    qmax_kernel<<<512, 256, 0, stream>>>(x, qkv_w, ln1_g, ln1_b, qattn_w, l, acc);
    qsum_kernel<<<512, 256, 0, stream>>>(x, qkv_w, ln1_g, ln1_b, qattn_w, l, acc);
    finalize_gq_kernel<<<1, 64, 0, stream>>>(acc);
    kmax_kernel<<<512, 256, 0, stream>>>(x, qkv_w, ln1_g, ln1_b, kattn_w, l, acc);
    ksum_kernel<<<512, 256, 0, stream>>>(x, qkv_w, ln1_g, ln1_b, kattn_w, l, acc);
    finalize_gk_kernel<<<1, 64, 0, stream>>>(acc);
    update_kernel<<<tokBlocks, 256, 0, stream>>>(x, qkv_w, ln1_g, ln1_b, tor_w, tor_b,
                                                 out_w, out_b, ln2_g, ln2_b,
                                                 ff1_w, ff1_b, ff2_w, ff2_b, l, acc);
  }

  rc_kernel<<<tokBlocks, 256, 0, stream>>>(x, corr_w, corr_b, rc30);
  attn_softmax_kernel<<<3600, 256, 0, stream>>>(rc30, attn);
  vh_kernel<<<(256 * 3616 + 255) / 256, 256, 0, stream>>>(v, vh);
  wv_wmma_kernel<<<113, 256, 0, stream>>>(attn, vh, wv_out);
  final_kernel<<<3600, 256, 0, stream>>>(f_q, wv_out, fq_out);
}
